// ModularWhisperAttention_74586402063325
// MI455X (gfx1250) — matrix operand order repaired and checked by reading, not compiled, not run on hardware
//
#include <hip/hip_runtime.h>

#define E   1280
#define S   1500
#define SP  1504   // S padded to multiple of 32
#define H   20
#define HD  64

typedef __attribute__((ext_vector_type(16))) __bf16 v16bf;
typedef __attribute__((ext_vector_type(8)))  __bf16 v8bf;
typedef __attribute__((ext_vector_type(8)))  float  v8f;

// ---- helpers -------------------------------------------------------------

__device__ __forceinline__ __bf16 f2bf(float f) {
  union { float f; unsigned u; } x; x.f = f;
  unsigned r = (x.u + 0x7FFFu + ((x.u >> 16) & 1u)) >> 16;   // RNE
  union { unsigned short s; __bf16 b; } y; y.s = (unsigned short)r;
  return y.b;
}

// A-matrix 16x32 bf16 fragment (row-major source, leading dim lda).
// Lane (half = lane>>4, m = lane&15): element s holds K = half*16 + s
// -> one contiguous 32B chunk.
__device__ __forceinline__ v16bf load_A(const __bf16* base, int lda) {
  int lane = threadIdx.x & 31;
  int half = lane >> 4;
  int m    = lane & 15;
  return *(const v16bf*)(base + m * lda + half * 16);
}

// B-matrix 32x16 bf16 fragment. Source layout: B[k][n] = M[n*ldb + k]
// (row-major [N][K] like torch Linear weights -> computes x @ W^T).
// Lane (half, n): VGPR v holds K = half*16 + 2v, 2v+1 -> one contiguous 32B.
__device__ __forceinline__ v16bf load_B(const __bf16* base, int ldb) {
  int lane = threadIdx.x & 31;
  int half = lane >> 4;
  int n    = lane & 15;
  return *(const v16bf*)(base + n * ldb + half * 16);
}

__device__ __forceinline__ v8f wmma_bf16(v16bf a, v16bf b, v8f c) {
  return __builtin_amdgcn_wmma_f32_16x16x32_bf16(
      /*neg_a=*/false, a, /*neg_b=*/false, b,
      /*c_mod=*/(short)0, c, /*reuse_a=*/false, /*reuse_b=*/false);
}

// Async copy 16B/lane from global memory into LDS (ASYNCcnt-tracked).
__device__ __forceinline__ void async_load_b128(unsigned lds_addr, const void* gaddr) {
  asm volatile("global_load_async_to_lds_b128 %0, %1, off"
               :: "v"(lds_addr), "v"((unsigned long long)(uintptr_t)gaddr)
               : "memory");
}

// Generic (shared) pointer -> LDS byte offset: shared aperture keeps the
// LDS offset in the low 32 bits of the generic address.
__device__ __forceinline__ unsigned lds_off(const void* p) {
  return (unsigned)(uintptr_t)p;
}

// ---- conversion kernels --------------------------------------------------

__global__ void k_cvt_x(const float* __restrict__ x, __bf16* __restrict__ xb) {
  int idx = blockIdx.x * blockDim.x + threadIdx.x;
  if (idx >= SP * E) return;
  int row = idx / E;
  float v = (row < S) ? x[idx] : 0.0f;
  xb[idx] = f2bf(v);
}

__global__ void k_cvt_w(const float* __restrict__ w, __bf16* __restrict__ wb, int n) {
  int idx = blockIdx.x * blockDim.x + threadIdx.x;
  if (idx < n) wb[idx] = f2bf(w[idx]);
}

// ---- projection GEMM: out = (X @ W^T + bias) * scale --------------------
// Block = 128 threads = 4 waves, block tile 32(M) x 256(N);
// each wave owns a 32x64 tile = 2x4 WMMA accumulators (8 WMMA / k-step).
// transposed==0: out[row*E + col] (Q, K). transposed==1: out[col*SP + row] (V^T).

__global__ void k_gemm_qkv(const __bf16* __restrict__ X, const __bf16* __restrict__ W,
                           const float* __restrict__ bias, float scale,
                           __bf16* __restrict__ out, int transposed) {
  int wid  = threadIdx.x >> 5;
  int lane = threadIdx.x & 31;
  int half = lane >> 4, ln = lane & 15;
  int m0 = blockIdx.x * 32;
  int n0 = blockIdx.y * 256 + wid * 64;

  v8f acc[2][4];
  v8f z = {};
#pragma unroll
  for (int mi = 0; mi < 2; ++mi)
#pragma unroll
    for (int j = 0; j < 4; ++j) acc[mi][j] = z;

  for (int k0 = 0; k0 < E; k0 += 32) {
    v16bf a0 = load_A(X + m0 * E + k0, E);
    v16bf a1 = load_A(X + (m0 + 16) * E + k0, E);
#pragma unroll
    for (int j = 0; j < 4; ++j) {
      v16bf b = load_B(W + (n0 + 16 * j) * E + k0, E);
      acc[0][j] = wmma_bf16(a0, b, acc[0][j]);
      acc[1][j] = wmma_bf16(a1, b, acc[1][j]);
    }
  }

#pragma unroll
  for (int mi = 0; mi < 2; ++mi) {
#pragma unroll
    for (int j = 0; j < 4; ++j) {
      int col = n0 + j * 16 + ln;
      float bb = bias ? bias[col] : 0.0f;
#pragma unroll
      for (int v = 0; v < 8; ++v) {
        int row = m0 + mi * 16 + half * 8 + v;
        float r = (acc[mi][j][v] + bb) * scale;
        if (!transposed) out[row * E + col] = f2bf(r);
        else             out[col * SP + row] = f2bf(r);
      }
    }
  }
}

// ---- flash attention -----------------------------------------------------
// Block = 64 threads = 2 waves; each wave owns 16 query rows of one head.
// K/V tiles (32 keys) are staged into LDS with async b128 copies, double
// buffered; the two waves split the staging and share the tiles via the
// workgroup split-barrier. Online softmax in fp32.

__global__ void k_attn(const __bf16* __restrict__ Q, const __bf16* __restrict__ Kb,
                       const __bf16* __restrict__ Vt, __bf16* __restrict__ O) {
  __shared__ __bf16 Klds[2][32 * 64];  // [t(32)][d(64)] per buffer
  __shared__ __bf16 Vlds[2][64 * 32];  // [d(64)][t(32)] per buffer
  __shared__ __bf16 Plds[2][16 * 32];  // per-wave P tile

  int wid  = threadIdx.x >> 5;
  int lane = threadIdx.x & 31;
  int half = lane >> 4, ln = lane & 15;
  int m0 = (blockIdx.x * 2 + wid) * 16;
  int h  = blockIdx.y;

  // stage one 32-key (K,V) tile; each wave issues 8 async b128 (4 K + 4 V)
  auto stage = [&](int t0, int buf) {
    unsigned kbase = lds_off(&Klds[buf][0]);
    const char* krow = (const char*)(Kb + (size_t)(t0 + lane) * E + h * HD);
#pragma unroll
    for (int i = 0; i < 4; ++i) {
      int c = wid * 4 + i;                       // 16B chunk within 128B row
      async_load_b128(kbase + lane * 128 + c * 16, krow + c * 16);
    }
    unsigned vbase = lds_off(&Vlds[buf][0]);
    int vr = lane >> 2, vc = lane & 3;
#pragma unroll
    for (int i = 0; i < 4; ++i) {
      int row = (wid * 4 + i) * 8 + vr;          // d-row 0..63
      const char* vrow = (const char*)(Vt + (size_t)(h * HD + row) * SP + t0);
      async_load_b128(vbase + row * 64 + vc * 16, vrow + vc * 16);
    }
  };

  const __bf16* qb = Q + (size_t)m0 * E + h * HD;
  v16bf aq0 = load_A(qb, E);        // d = 0..31
  v16bf aq1 = load_A(qb + 32, E);   // d = 32..63

  v8f o0 = {}, o1 = {}, o2 = {}, o3 = {};
  float rmax[8], rsum[8];
#pragma unroll
  for (int v = 0; v < 8; ++v) { rmax[v] = -3.0e38f; rsum[v] = 0.0f; }

  stage(0, 0);

  for (int t0 = 0; t0 < SP; t0 += 32) {
    int cur = (t0 >> 5) & 1;
    if (t0 + 32 < SP) {
      stage(t0 + 32, cur ^ 1);
      asm volatile("s_wait_asynccnt 0x8" ::: "memory");  // current tile done
    } else {
      asm volatile("s_wait_asynccnt 0x0" ::: "memory");
    }
    __syncthreads();   // both waves' halves of the tile visible

    const __bf16* Kt = &Klds[cur][0];
    const __bf16* Vc = &Vlds[cur][0];

    // ---- scores: two 16x16 tiles over keys [t0, t0+32), K-depth 64
    v8f s[2];
#pragma unroll
    for (int jt = 0; jt < 2; ++jt) {
      const __bf16* kb = Kt + jt * 16 * 64;
      v8f zz = {};
      s[jt] = wmma_bf16(aq0, load_B(kb, 64), zz);
      s[jt] = wmma_bf16(aq1, load_B(kb + 32, 64), s[jt]);
      int key = t0 + jt * 16 + ln;
      if (key >= S) {
#pragma unroll
        for (int v = 0; v < 8; ++v) s[jt][v] = -1.0e30f;
      }
    }

    // ---- online softmax (rows in VGPR index v, replicated over 16 lanes)
    float p0[8], p1[8];
#pragma unroll
    for (int v = 0; v < 8; ++v) {
      float mv = fmaxf(s[0][v], s[1][v]);
      mv = fmaxf(mv, __shfl_xor(mv, 1, 32));
      mv = fmaxf(mv, __shfl_xor(mv, 2, 32));
      mv = fmaxf(mv, __shfl_xor(mv, 4, 32));
      mv = fmaxf(mv, __shfl_xor(mv, 8, 32));
      float mnew = fmaxf(rmax[v], mv);
      float corr = __expf(rmax[v] - mnew);
      rmax[v] = mnew;
      rsum[v] *= corr;
      o0[v] *= corr; o1[v] *= corr; o2[v] *= corr; o3[v] *= corr;
      p0[v] = __expf(s[0][v] - mnew);
      p1[v] = __expf(s[1][v] - mnew);
      rsum[v] += p0[v] + p1[v];   // per-lane partial; cross-lane reduce at end
      Plds[wid][(half * 8 + v) * 32 + ln]      = f2bf(p0[v]);
      Plds[wid][(half * 8 + v) * 32 + 16 + ln] = f2bf(p1[v]);
    }
    asm volatile("s_wait_dscnt 0" ::: "memory");

    // ---- O += P @ V  (A = P 16x32 from LDS, B = V tile slices from LDS)
    v16bf ap = load_A(&Plds[wid][0], 32);
    o0 = wmma_bf16(ap, load_B(Vc,           32), o0);
    o1 = wmma_bf16(ap, load_B(Vc + 16 * 32, 32), o1);
    o2 = wmma_bf16(ap, load_B(Vc + 32 * 32, 32), o2);
    o3 = wmma_bf16(ap, load_B(Vc + 48 * 32, 32), o3);

    __syncthreads();   // protect buffer cur before it is restaged
  }

  // ---- finalize: reduce row sums across lanes, normalize, store
  float inv[8];
#pragma unroll
  for (int v = 0; v < 8; ++v) {
    float r = rsum[v];
    r += __shfl_xor(r, 1, 32);
    r += __shfl_xor(r, 2, 32);
    r += __shfl_xor(r, 4, 32);
    r += __shfl_xor(r, 8, 32);
    inv[v] = 1.0f / r;
  }
#pragma unroll
  for (int dj = 0; dj < 4; ++dj) {
    v8f a = (dj == 0) ? o0 : (dj == 1) ? o1 : (dj == 2) ? o2 : o3;
#pragma unroll
    for (int v = 0; v < 8; ++v) {
      int row = m0 + half * 8 + v;
      O[row * E + h * HD + dj * 16 + ln] = f2bf(a[v] * inv[v]);
    }
  }
}

// ---- output projection: out = Ob @ Wo^T + bo (fp32 result) ---------------

__global__ void k_gemm_out(const __bf16* __restrict__ A, const __bf16* __restrict__ W,
                           const float* __restrict__ bias, float* __restrict__ out) {
  int wid  = threadIdx.x >> 5;
  int lane = threadIdx.x & 31;
  int half = lane >> 4, ln = lane & 15;
  int m0 = blockIdx.x * 32;
  int n0 = blockIdx.y * 256 + wid * 64;

  v8f acc[2][4];
  v8f z = {};
#pragma unroll
  for (int mi = 0; mi < 2; ++mi)
#pragma unroll
    for (int j = 0; j < 4; ++j) acc[mi][j] = z;

  for (int k0 = 0; k0 < E; k0 += 32) {
    v16bf a0 = load_A(A + m0 * E + k0, E);
    v16bf a1 = load_A(A + (m0 + 16) * E + k0, E);
#pragma unroll
    for (int j = 0; j < 4; ++j) {
      v16bf b = load_B(W + (n0 + 16 * j) * E + k0, E);
      acc[0][j] = wmma_bf16(a0, b, acc[0][j]);
      acc[1][j] = wmma_bf16(a1, b, acc[1][j]);
    }
  }

#pragma unroll
  for (int mi = 0; mi < 2; ++mi) {
#pragma unroll
    for (int j = 0; j < 4; ++j) {
      int col = n0 + j * 16 + ln;
      float bb = bias[col];
#pragma unroll
      for (int v = 0; v < 8; ++v) {
        int row = m0 + mi * 16 + half * 8 + v;
        if (row < S) out[row * E + col] = acc[mi][j][v] + bb;
      }
    }
  }
}

// ---- host launcher -------------------------------------------------------

extern "C" void kernel_launch(void* const* d_in, const int* in_sizes, int n_in,
                              void* d_out, int out_size, void* d_ws, size_t ws_size,
                              hipStream_t stream) {
  (void)in_sizes; (void)n_in; (void)out_size; (void)ws_size;
  const float* x  = (const float*)d_in[0];
  const float* Wq = (const float*)d_in[1];
  const float* bq = (const float*)d_in[2];
  const float* Wk = (const float*)d_in[3];
  const float* Wv = (const float*)d_in[4];
  const float* bv = (const float*)d_in[5];
  const float* Wo = (const float*)d_in[6];
  const float* bo = (const float*)d_in[7];
  float* out = (float*)d_out;

  char* ws = (char*)d_ws;
  size_t off = 0;
  auto alloc = [&](size_t bytes) {
    void* p = ws + off;
    off += (bytes + 255) & ~(size_t)255;
    return p;
  };
  const size_t actB = (size_t)SP * E * sizeof(__bf16);
  const size_t wB   = (size_t)E * E * sizeof(__bf16);
  __bf16* xb   = (__bf16*)alloc(actB);
  __bf16* wqb  = (__bf16*)alloc(wB);
  __bf16* wkb  = (__bf16*)alloc(wB);
  __bf16* wvb  = (__bf16*)alloc(wB);
  __bf16* wob  = (__bf16*)alloc(wB);
  __bf16* Qb   = (__bf16*)alloc(actB);
  __bf16* Kbuf = (__bf16*)alloc(actB);
  __bf16* Vt   = (__bf16*)alloc(actB);   // V transposed: [E][SP]
  __bf16* Ob   = (__bf16*)alloc(actB);

  // 1) bf16 conversions
  {
    int n = SP * E;
    k_cvt_x<<<(n + 255) / 256, 256, 0, stream>>>(x, xb);
    int nw = E * E;
    int g = (nw + 255) / 256;
    k_cvt_w<<<g, 256, 0, stream>>>(Wq, wqb, nw);
    k_cvt_w<<<g, 256, 0, stream>>>(Wk, wkb, nw);
    k_cvt_w<<<g, 256, 0, stream>>>(Wv, wvb, nw);
    k_cvt_w<<<g, 256, 0, stream>>>(Wo, wob, nw);
  }

  // 2) QKV projections (Q pre-scaled by 1/sqrt(64); V stored transposed)
  dim3 gg(SP / 32, E / 256);   // 47 x 5
  k_gemm_qkv<<<gg, 128, 0, stream>>>(xb, wqb, bq, 0.125f, Qb, 0);
  k_gemm_qkv<<<gg, 128, 0, stream>>>(xb, wkb, nullptr, 1.0f, Kbuf, 0);
  k_gemm_qkv<<<gg, 128, 0, stream>>>(xb, wvb, bv, 1.0f, Vt, 1);

  // 3) flash attention: 2 waves/block share async-staged K/V tiles
  dim3 ga(SP / 32, H);         // 47 x 20 (2 query tiles per block)
  k_attn<<<ga, 64, 0, stream>>>(Qb, Kbuf, Vt, Ob);

  // 4) output projection (fp32 out)
  k_gemm_out<<<gg, 128, 0, stream>>>(Ob, wob, bo, out);
}
